// GCNLayer_21835613732936
// MI455X (gfx1250) — compile-verified, matrix operand order checked by reading, not run on hardware
//
#include <hip/hip_runtime.h>
#include <stdint.h>
#include <stddef.h>

typedef __bf16 bf16;
typedef __attribute__((ext_vector_type(8)))  bf16  v8bf;
typedef __attribute__((ext_vector_type(16))) bf16  v16bf;
typedef __attribute__((ext_vector_type(8)))  float v8f;

#define NN   8192
#define DIM  256
#define BM   64
#define BN   128
#define BK   32

#if defined(__gfx1250__) && \
    __has_builtin(__builtin_amdgcn_global_load_async_to_lds_b128) && \
    __has_builtin(__builtin_amdgcn_s_wait_asynccnt)
#define USE_ASYNC_LDS 1
// builtin signature (from clang diagnostic): parameters are pointers to
// gcc-style int4 vectors in addrspace(1) (global) / addrspace(3) (LDS).
typedef int v4i_g __attribute__((vector_size(16)));
typedef __attribute__((address_space(1))) v4i_g as1_v4i;
typedef __attribute__((address_space(3))) v4i_g as3_v4i;
#define TO_AS1(p) ((as1_v4i*)(uintptr_t)(p))
#define TO_AS3(p) ((as3_v4i*)(uint32_t)(uintptr_t)(p))
#else
#define USE_ASYNC_LDS 0
#endif

#define SHUF16(lo8, hi8) __builtin_shufflevector(lo8, hi8, \
    0,1,2,3,4,5,6,7,8,9,10,11,12,13,14,15)

// ---------------------------------------------------------------------------
// Kernel 1: degree[i] = 1 + sum_j adj[i][j]; dis[i] = rsqrt(degree[i])
// ---------------------------------------------------------------------------
__global__ __launch_bounds__(256) void k_degree(const float* __restrict__ adj,
                                                float* __restrict__ dis) {
    const int wave = threadIdx.x >> 5;
    const int lane = threadIdx.x & 31;
    const int row  = blockIdx.x * 8 + wave;
    const float* p = adj + (size_t)row * NN;
    float s = 0.0f;
#pragma unroll 4
    for (int j = 0; j < 64; ++j) {
        float4 v = *(const float4*)(p + j * 128 + lane * 4);
        s += v.x + v.y + v.z + v.w;
    }
#pragma unroll
    for (int off = 16; off > 0; off >>= 1) s += __shfl_xor(s, off, 32);
    if (lane == 0) dis[row] = rsqrtf(s + 1.0f);
}

// ---------------------------------------------------------------------------
// Kernel 2: xsT[n][k] = bf16( dis[k] * x[k][n] )   (transposed, k-contiguous)
// ---------------------------------------------------------------------------
__global__ __launch_bounds__(256) void k_scale_x(const float* __restrict__ x,
                                                 const float* __restrict__ dis,
                                                 bf16* __restrict__ xsT) {
    const int tid = blockIdx.x * 256 + threadIdx.x;
    const int k = tid & (NN - 1);
    const int n = tid >> 13;
    xsT[(size_t)n * NN + k] = (bf16)(dis[k] * x[(size_t)k * DIM + n]);
}

// ---------------------------------------------------------------------------
// Kernel 3: WT[o][i] = bf16( W[i][o] )
// ---------------------------------------------------------------------------
__global__ __launch_bounds__(256) void k_wt(const float* __restrict__ W,
                                            bf16* __restrict__ WT) {
    const int i = blockIdx.x;
    const int o = threadIdx.x;
    WT[(size_t)o * DIM + i] = (bf16)W[(size_t)i * DIM + o];
}

// ---------------------------------------------------------------------------
// Kernel 4: Y = dis[m] * ( adj @ xs + xs[m,:] ), stored bf16 row-major.
// BM=64 x BN=128 per 4-wave WG; each wave owns a 32x64 (2x4 tile) patch.
// Double-buffered LDS; B tile staged via async global->LDS when available.
// ---------------------------------------------------------------------------
__global__ __launch_bounds__(128) void k_agg(const float* __restrict__ adj,
                                             const bf16* __restrict__ xsT,
                                             const float* __restrict__ dis,
                                             bf16* __restrict__ Ybf) {
    __shared__ __align__(16) bf16 As[2][BM][BK];
    __shared__ __align__(16) bf16 Bs[2][BN][BK];

    const int tid  = threadIdx.x;
    const int wave = tid >> 5;
    const int lane = tid & 31;
    const int lo   = lane & 15;
    const int hi   = lane >> 4;
    const int wm   = wave & 1;   // M half (2 tiles of 16 rows)
    const int wn   = wave >> 1;  // N half (4 tiles of 16 cols)
    const int m0 = blockIdx.x * BM;
    const int n1 = blockIdx.y * BN;

    const int lr    = tid >> 1;  // A staging row (0..63)
    const int lhalf = tid & 1;   // 16-float half of the BK row

    // ISA 16-bit A-matrix layout: lanes 0-15 hold K 0..7 & 16..23,
    // lanes 16-31 hold K 8..15 & 24..31.
    const int c0 = 8 * hi;
    const int c1 = 16 + 8 * hi;
    const int amrow = wm * 32 + lo;
    const int bnbase = wn * 64;

    v8f acc[2][4] = {};

    // ---- prologue: stage tile k0=0 into buffer 0 ----
    {
        const float* srcA = adj + (size_t)(m0 + lr) * NN + lhalf * 16;
        float4 f0 = ((const float4*)srcA)[0];
        float4 f1 = ((const float4*)srcA)[1];
        float4 f2 = ((const float4*)srcA)[2];
        float4 f3 = ((const float4*)srcA)[3];
        v8bf b0, b1;
        b0[0]=(bf16)f0.x; b0[1]=(bf16)f0.y; b0[2]=(bf16)f0.z; b0[3]=(bf16)f0.w;
        b0[4]=(bf16)f1.x; b0[5]=(bf16)f1.y; b0[6]=(bf16)f1.z; b0[7]=(bf16)f1.w;
        b1[0]=(bf16)f2.x; b1[1]=(bf16)f2.y; b1[2]=(bf16)f2.z; b1[3]=(bf16)f2.w;
        b1[4]=(bf16)f3.x; b1[5]=(bf16)f3.y; b1[6]=(bf16)f3.z; b1[7]=(bf16)f3.w;
        *(v8bf*)&As[0][lr][lhalf * 16]     = b0;
        *(v8bf*)&As[0][lr][lhalf * 16 + 8] = b1;

        const bf16* gB = xsT + (size_t)(n1 + tid) * NN;
#if USE_ASYNC_LDS
        __builtin_amdgcn_global_load_async_to_lds_b128(TO_AS1(gB), TO_AS3(&Bs[0][tid][0]), 0,  0);
        __builtin_amdgcn_global_load_async_to_lds_b128(TO_AS1(gB), TO_AS3(&Bs[0][tid][0]), 16, 0);
        __builtin_amdgcn_global_load_async_to_lds_b128(TO_AS1(gB), TO_AS3(&Bs[0][tid][0]), 32, 0);
        __builtin_amdgcn_global_load_async_to_lds_b128(TO_AS1(gB), TO_AS3(&Bs[0][tid][0]), 48, 0);
        __builtin_amdgcn_s_wait_asynccnt(0);
#else
        v8bf t0 = ((const v8bf*)gB)[0];
        v8bf t1 = ((const v8bf*)gB)[1];
        v8bf t2 = ((const v8bf*)gB)[2];
        v8bf t3 = ((const v8bf*)gB)[3];
        *(v8bf*)&Bs[0][tid][0]  = t0;
        *(v8bf*)&Bs[0][tid][8]  = t1;
        *(v8bf*)&Bs[0][tid][16] = t2;
        *(v8bf*)&Bs[0][tid][24] = t3;
#endif
    }
    __syncthreads();

    for (int k0 = 0; k0 < NN; k0 += BK) {
        const int buf = (k0 >> 5) & 1;
        const bool has_next = (k0 + BK) < NN;

        // ---- issue next-tile global loads early (overlap with WMMA) ----
        float4 f0, f1, f2, f3;
        if (has_next) {
            const float* srcA = adj + (size_t)(m0 + lr) * NN + (k0 + BK) + lhalf * 16;
            f0 = ((const float4*)srcA)[0];
            f1 = ((const float4*)srcA)[1];
            f2 = ((const float4*)srcA)[2];
            f3 = ((const float4*)srcA)[3];
#if USE_ASYNC_LDS
            const bf16* gB = xsT + (size_t)(n1 + tid) * NN + (k0 + BK);
            as3_v4i* lB = TO_AS3(&Bs[buf ^ 1][tid][0]);
            __builtin_amdgcn_global_load_async_to_lds_b128(TO_AS1(gB), lB, 0,  0);
            __builtin_amdgcn_global_load_async_to_lds_b128(TO_AS1(gB), lB, 16, 0);
            __builtin_amdgcn_global_load_async_to_lds_b128(TO_AS1(gB), lB, 32, 0);
            __builtin_amdgcn_global_load_async_to_lds_b128(TO_AS1(gB), lB, 48, 0);
#endif
            if (k0 + 2 * BK < NN)
                __builtin_prefetch(adj + (size_t)(m0 + lr) * NN + (k0 + 2 * BK) + lhalf * 16, 0, 1);
        }
#if !USE_ASYNC_LDS
        v8bf t0, t1, t2, t3;
        if (has_next) {
            const bf16* gB = xsT + (size_t)(n1 + tid) * NN + (k0 + BK);
            t0 = ((const v8bf*)gB)[0];
            t1 = ((const v8bf*)gB)[1];
            t2 = ((const v8bf*)gB)[2];
            t3 = ((const v8bf*)gB)[3];
        }
#endif

        // ---- compute on current buffer: 6 frag loads, 8 WMMAs ----
        v8bf aL0 = *(const v8bf*)&As[buf][amrow][c0];
        v8bf aH0 = *(const v8bf*)&As[buf][amrow][c1];
        v8bf aL1 = *(const v8bf*)&As[buf][amrow + 16][c0];
        v8bf aH1 = *(const v8bf*)&As[buf][amrow + 16][c1];
        v16bf af0 = SHUF16(aL0, aH0);
        v16bf af1 = SHUF16(aL1, aH1);

#pragma unroll
        for (int nt = 0; nt < 4; ++nt) {
            const int bcol = bnbase + nt * 16 + lo;
            v8bf bLo = *(const v8bf*)&Bs[buf][bcol][c0];
            v8bf bHi = *(const v8bf*)&Bs[buf][bcol][c1];
            v16bf bfrag = SHUF16(bLo, bHi);
            acc[0][nt] = __builtin_amdgcn_wmma_f32_16x16x32_bf16(
                false, af0, false, bfrag, (short)0, acc[0][nt], false, false);
            acc[1][nt] = __builtin_amdgcn_wmma_f32_16x16x32_bf16(
                false, af1, false, bfrag, (short)0, acc[1][nt], false, false);
        }

        // ---- convert + store next A tile (and B tile in fallback) ----
        if (has_next) {
            v8bf b0, b1;
            b0[0]=(bf16)f0.x; b0[1]=(bf16)f0.y; b0[2]=(bf16)f0.z; b0[3]=(bf16)f0.w;
            b0[4]=(bf16)f1.x; b0[5]=(bf16)f1.y; b0[6]=(bf16)f1.z; b0[7]=(bf16)f1.w;
            b1[0]=(bf16)f2.x; b1[1]=(bf16)f2.y; b1[2]=(bf16)f2.z; b1[3]=(bf16)f2.w;
            b1[4]=(bf16)f3.x; b1[5]=(bf16)f3.y; b1[6]=(bf16)f3.z; b1[7]=(bf16)f3.w;
            *(v8bf*)&As[buf ^ 1][lr][lhalf * 16]     = b0;
            *(v8bf*)&As[buf ^ 1][lr][lhalf * 16 + 8] = b1;
#if !USE_ASYNC_LDS
            *(v8bf*)&Bs[buf ^ 1][tid][0]  = t0;
            *(v8bf*)&Bs[buf ^ 1][tid][8]  = t1;
            *(v8bf*)&Bs[buf ^ 1][tid][16] = t2;
            *(v8bf*)&Bs[buf ^ 1][tid][24] = t3;
#endif
        }
#if USE_ASYNC_LDS
        __builtin_amdgcn_s_wait_asynccnt(0);
#endif
        __syncthreads();
    }

    // ---- epilogue: Y[m][n] = dis[m]*(T + xs[m][n]), stored bf16 ----
    float dm[2][8];
#pragma unroll
    for (int mt = 0; mt < 2; ++mt)
#pragma unroll
        for (int v = 0; v < 8; ++v)
            dm[mt][v] = dis[m0 + wm * 32 + mt * 16 + v + 8 * hi];

#pragma unroll
    for (int mt = 0; mt < 2; ++mt) {
#pragma unroll
        for (int nt = 0; nt < 4; ++nt) {
            const int n = n1 + bnbase + nt * 16 + lo;
#pragma unroll
            for (int v = 0; v < 8; ++v) {
                const int m = m0 + wm * 32 + mt * 16 + v + 8 * hi;
                float xv = (float)xsT[(size_t)n * NN + m];   // xs[m][n]
                float y  = dm[mt][v] * (acc[mt][nt][v] + xv);
                Ybf[(size_t)m * DIM + n] = (bf16)y;
            }
        }
    }
}

// ---------------------------------------------------------------------------
// Kernel 5: out = ReLU( Y @ W + b ), fp32 output. K = 256 (8 iterations).
// ---------------------------------------------------------------------------
__global__ __launch_bounds__(128) void k_out(const bf16* __restrict__ Ybf,
                                             const bf16* __restrict__ WT,
                                             const float* __restrict__ bias,
                                             float* __restrict__ out) {
    __shared__ __align__(16) bf16 As[BM][BK];
    __shared__ __align__(16) bf16 Bs[BN][BK];

    const int tid  = threadIdx.x;
    const int wave = tid >> 5;
    const int lane = tid & 31;
    const int lo   = lane & 15;
    const int hi   = lane >> 4;
    const int wm   = wave & 1;
    const int wn   = wave >> 1;
    const int m0 = blockIdx.x * BM;
    const int n1 = blockIdx.y * BN;

    const int lr    = tid >> 1;
    const int lhalf = tid & 1;

    v8f acc[2][4] = {};
    const int c0 = 8 * hi;
    const int c1 = 16 + 8 * hi;
    const int amrow = wm * 32 + lo;
    const int bnbase = wn * 64;

    for (int k0 = 0; k0 < DIM; k0 += BK) {
        {
            const bf16* src = Ybf + (size_t)(m0 + lr) * DIM + k0 + lhalf * 16;
            v8bf a0 = ((const v8bf*)src)[0];
            v8bf a1 = ((const v8bf*)src)[1];
            *(v8bf*)&As[lr][lhalf * 16]     = a0;
            *(v8bf*)&As[lr][lhalf * 16 + 8] = a1;
        }
        {
            const bf16* src = WT + (size_t)(n1 + tid) * DIM + k0;
            v8bf t0 = ((const v8bf*)src)[0];
            v8bf t1 = ((const v8bf*)src)[1];
            v8bf t2 = ((const v8bf*)src)[2];
            v8bf t3 = ((const v8bf*)src)[3];
            *(v8bf*)&Bs[tid][0]  = t0;
            *(v8bf*)&Bs[tid][8]  = t1;
            *(v8bf*)&Bs[tid][16] = t2;
            *(v8bf*)&Bs[tid][24] = t3;
        }
        __syncthreads();

        v8bf aL0 = *(const v8bf*)&As[amrow][c0];
        v8bf aH0 = *(const v8bf*)&As[amrow][c1];
        v8bf aL1 = *(const v8bf*)&As[amrow + 16][c0];
        v8bf aH1 = *(const v8bf*)&As[amrow + 16][c1];
        v16bf af0 = SHUF16(aL0, aH0);
        v16bf af1 = SHUF16(aL1, aH1);

#pragma unroll
        for (int nt = 0; nt < 4; ++nt) {
            const int bcol = bnbase + nt * 16 + lo;
            v8bf bLo = *(const v8bf*)&Bs[bcol][c0];
            v8bf bHi = *(const v8bf*)&Bs[bcol][c1];
            v16bf bfrag = SHUF16(bLo, bHi);
            acc[0][nt] = __builtin_amdgcn_wmma_f32_16x16x32_bf16(
                false, af0, false, bfrag, (short)0, acc[0][nt], false, false);
            acc[1][nt] = __builtin_amdgcn_wmma_f32_16x16x32_bf16(
                false, af1, false, bfrag, (short)0, acc[1][nt], false, false);
        }
        __syncthreads();
    }

#pragma unroll
    for (int mt = 0; mt < 2; ++mt) {
#pragma unroll
        for (int nt = 0; nt < 4; ++nt) {
            const int n = n1 + bnbase + nt * 16 + lo;
            const float bn = bias[n];
#pragma unroll
            for (int v = 0; v < 8; ++v) {
                const int m = m0 + wm * 32 + mt * 16 + v + 8 * hi;
                float z = acc[mt][nt][v] + bn;
                out[(size_t)m * DIM + n] = fmaxf(z, 0.0f);
            }
        }
    }
}

// ---------------------------------------------------------------------------
extern "C" void kernel_launch(void* const* d_in, const int* in_sizes, int n_in,
                              void* d_out, int out_size, void* d_ws, size_t ws_size,
                              hipStream_t stream) {
    const float* x   = (const float*)d_in[0];   // [8192, 256]
    const float* adj = (const float*)d_in[1];   // [8192, 8192]
    const float* W   = (const float*)d_in[2];   // [256, 256]
    const float* b   = (const float*)d_in[3];   // [256]
    float* out = (float*)d_out;                 // [8192, 256]

    char* ws = (char*)d_ws;
    float* dis = (float*)ws;                                            // 32 KB
    bf16*  xsT = (bf16*)(ws + 32 * 1024);                               // 4 MB
    bf16*  WT  = (bf16*)(ws + 32 * 1024 + (size_t)NN * DIM * 2);        // 128 KB
    bf16*  Ybf = (bf16*)(ws + 32 * 1024 + (size_t)NN * DIM * 2
                              + (size_t)DIM * DIM * 2);                 // 4 MB

    k_degree <<<NN / 8, 256, 0, stream>>>(adj, dis);
    k_scale_x<<<(NN * DIM) / 256, 256, 0, stream>>>(x, dis, xsT);
    k_wt     <<<DIM, DIM, 0, stream>>>(W, WT);

    dim3 g(NN / BM, DIM / BN);   // 128 x 2
    k_agg<<<g, 128, 0, stream>>>(adj, xsT, dis, Ybf);
    k_out<<<g, 128, 0, stream>>>(Ybf, WT, b, out);
}
